// RNNModel_26706106647026
// MI455X (gfx1250) — matrix operand order repaired and checked by reading, not compiled, not run on hardware
//
#include <hip/hip_runtime.h>
#include <math.h>

// ---------- CDNA5 WMMA types ----------
typedef __attribute__((ext_vector_type(16))) __bf16 v16bf;
typedef __attribute__((ext_vector_type(8)))  __bf16 v8bf;
typedef __attribute__((ext_vector_type(8)))  float  v8f;
typedef __attribute__((ext_vector_type(4)))  int    v4i;
typedef __attribute__((ext_vector_type(4)))  unsigned int u32x4;
typedef __attribute__((ext_vector_type(4)))  int    i32x4;
typedef __attribute__((ext_vector_type(8)))  int    i32x8;

#define BATCH 128
#define SEQ   512
#define INDIM 768
#define HID   64

static __device__ __forceinline__ v16bf cat8(v8bf lo, v8bf hi) {
  return __builtin_shufflevector(lo, hi, 0,1,2,3,4,5,6,7,8,9,10,11,12,13,14,15);
}

static __device__ __forceinline__ v8f wmma_bf16(v16bf a, v16bf b, v8f c) {
  // D = A(16x32 bf16) * B(32x16 bf16) + C(16x16 f32)
  return __builtin_amdgcn_wmma_f32_16x16x32_bf16(false, a, false, b, (short)0, c,
                                                 false, false);
}

// ---------- CDNA5 async global->LDS (ASYNCcnt-tracked), guarded ----------
#if defined(__AMDGCN__) && __has_builtin(__builtin_amdgcn_global_load_async_to_lds_b128)
#define USE_ASYNC 1
#else
#define USE_ASYNC 0
#endif

// ---------- CDNA5 Tensor Data Mover (TENSORcnt-tracked), guarded ----------
#if defined(__AMDGCN__) && __has_builtin(__builtin_amdgcn_tensor_load_to_lds)
#define USE_TDM 1
#else
#define USE_TDM 0
#endif

#define AS1 __attribute__((address_space(1)))
#define AS3 __attribute__((address_space(3)))

#if USE_ASYNC
static __device__ __forceinline__ void async_cp16(const float* g, float* l) {
  // builtin signature: (int4 AS1* src, int4 AS3* dst, imm offset, imm cpol)
  __builtin_amdgcn_global_load_async_to_lds_b128((AS1 v4i*)(v4i*)g,
                                                 (AS3 v4i*)(v4i*)l, 0, 0);
}
#endif

static __device__ __forceinline__ void wait_async() {
#if USE_ASYNC
#if __has_builtin(__builtin_amdgcn_s_wait_asynccnt)
  __builtin_amdgcn_s_wait_asynccnt(0);
#else
  asm volatile("s_wait_asynccnt 0x0" ::: "memory");
#endif
#endif
}

#if USE_TDM
// One TDM DMA of a 2D f32 tile (tile_d0 elems per row, tile_d1 rows, row
// stride `stride0` elems) from global to contiguous LDS. D# packed per ISA
// CDNA5 ISA (group0: count/lds_addr/global_addr/type=2; group1:
// data_size=4B, tensor dims/strides, tile dims). Issue from ONE wave only.
static __device__ __forceinline__ void tdm_load_2d(
    const float* gsrc, float* ldst,
    unsigned tile_d0, unsigned tile_d1, unsigned stride0)
{
  unsigned long long ga = (unsigned long long)(uintptr_t)gsrc;
  unsigned lds = (unsigned)(unsigned long long)(AS3 char*)ldst;
  const unsigned td0 = stride0;        // tensor_dim0 >= tile_d0 (no OOB)
  const unsigned td1 = 1u << 20;       // tensor_dim1: large, never OOB

  u32x4 g0 = {};
  g0[0] = 1u;                                                // count=1
  g0[1] = lds;                                               // lds_addr
  g0[2] = (unsigned)ga;                                      // gaddr[31:0]
  g0[3] = (unsigned)((ga >> 32) & 0x01FFFFFFu) | (2u << 30); // gaddr[56:32]|type=2

  i32x8 g1 = {};
  g1[0] = 0x00020000;                                        // data_size=2 (4B)
  g1[1] = (int)((td0 & 0xFFFFu) << 16);                      // tensor_dim0 lo
  g1[2] = (int)((td0 >> 16) | ((td1 & 0xFFFFu) << 16));      // dim0 hi | dim1 lo
  g1[3] = (int)(((td1 >> 16) & 0xFFFFu) | (tile_d0 << 16));  // dim1 hi | tile_dim0
  g1[4] = (int)tile_d1;                                      // tile_dim1 (tile_dim2=0)
  g1[5] = (int)stride0;                                      // tensor_dim0_stride lo
  g1[6] = 0;                                                 // stride0 hi | stride1 lo
  g1[7] = 0;

  i32x4 z4 = {};
#if __clang_major__ >= 23
  i32x8 z8 = {};
  __builtin_amdgcn_tensor_load_to_lds(g0, g1, z4, z4, z8, 0);
#else
  __builtin_amdgcn_tensor_load_to_lds(g0, g1, z4, z4, 0);
#endif
}
#endif

// Build bf16 fragments from f32 data staged in LDS (convert at frag time).
static __device__ __forceinline__ v16bf afrag_f32(const float* p, int kbase) {
  v16bf r;
#pragma unroll
  for (int i = 0; i < 8; ++i) {
    r[i]     = (__bf16)p[kbase + i];
    r[i + 8] = (__bf16)p[kbase + 8 + i];
  }
  return r;
}
static __device__ __forceinline__ v16bf bfrag_f32(const float* p) {
  v16bf r;
#pragma unroll
  for (int i = 0; i < 16; ++i) r[i] = (__bf16)p[i];
  return r;
}

// Stage x chunk (128x32 f32, row stride 768): TDM if available, else async.
static __device__ __forceinline__ void stage_x(
    const float* __restrict__ x, int row0, int k0, float* xbuf, int tid, int wave)
{
#if USE_TDM
  if (wave == 0)   // wave-uniform branch: TDM issues once (EXEC is ignored)
    tdm_load_2d(&x[(size_t)row0 * INDIM + k0], xbuf, 32, 128, INDIM);
#elif USE_ASYNC
#pragma unroll
  for (int c = 0; c < 4; ++c) {            // 128*32 f32 = 1024 x 16B
    int e = (c * 256 + tid) * 4;
    int r = e >> 5, k = e & 31;
    async_cp16(&x[(size_t)(row0 + r) * INDIM + k0 + k], &xbuf[e]);
  }
#else
#pragma unroll
  for (int c = 0; c < 4; ++c) {
    int e = (c * 256 + tid) * 4;
    int r = e >> 5, k = e & 31;
    *(float4*)&xbuf[e] = *(const float4*)&x[(size_t)(row0 + r) * INDIM + k0 + k];
  }
#endif
}

// Stage W chunk (64x32 f32) via async global->LDS (all threads).
static __device__ __forceinline__ void stage_w(
    const float* __restrict__ W, int k0, float* wbuf, int tid)
{
#if USE_ASYNC
#pragma unroll
  for (int c = 0; c < 2; ++c) {            // 64*32 f32 = 512 x 16B
    int e = (c * 256 + tid) * 4;
    int n = e >> 5, k = e & 31;
    async_cp16(&W[(size_t)n * INDIM + k0 + k], &wbuf[e]);
  }
#else
#pragma unroll
  for (int c = 0; c < 2; ++c) {
    int e = (c * 256 + tid) * 4;
    int n = e >> 5, k = e & 31;
    *(float4*)&wbuf[e] = *(const float4*)&W[(size_t)n * INDIM + k0 + k];
  }
#endif
}

static __device__ __forceinline__ void wait_stage(int wave) {
#if USE_TDM
  if (wave == 0) __builtin_amdgcn_s_wait_tensorcnt(0);
#endif
  wait_async();
}

// =====================================================================
// Kernel 1: P[t][b][n] = sum_i x[b,t,i]*W_ih0[n,i] + b_ih0[n] + b_hh0[n]
// 512 blocks x 256 threads. Double-buffered pipeline: TDM (x) + async (W)
// for chunk k+1 overlapped with bf16 WMMA on chunk k.
// =====================================================================
__global__ __launch_bounds__(256) void pre0_gemm_kernel(
    const float* __restrict__ x, const float* __restrict__ Wih,
    const float* __restrict__ bih, const float* __restrict__ bhh,
    float* __restrict__ P)
{
  __shared__ __align__(16) float xf[2][128 * 32];  // 2 x 16 KB
  __shared__ __align__(16) float wf[2][64 * 32];   // 2 x  8 KB

  const int tid   = threadIdx.x;
  const int lane  = tid & 31;
  const int wave  = tid >> 5;
  const int lrow  = lane & 15;
  const int kbase = (lane < 16) ? 0 : 16;   // A-fragment K base
  const int kb2   = (lane < 16) ? 0 : 16;   // B-fragment K base
  const int mhalf = (lane < 16) ? 0 : 8;    // C row half
  const int m0    = wave * 16;
  const int row0  = blockIdx.x * 128;

  float bsum[4];
#pragma unroll
  for (int j = 0; j < 4; ++j) {
    int n = j * 16 + lrow;
    bsum[j] = bih[n] + bhh[n];
  }

  stage_x(x, row0, 0, xf[0], tid, wave);
  stage_w(Wih, 0, wf[0], tid);

  v8f acc[4] = {};
  const int NIT = INDIM / 32;  // 24
  for (int it = 0; it < NIT; ++it) {
    wait_stage(wave);
    __syncthreads();                       // chunk `it` visible to all waves
    if (it + 1 < NIT) {                    // overlap next chunk with compute
      stage_x(x, row0, (it + 1) * 32, xf[(it + 1) & 1], tid, wave);
      stage_w(Wih, (it + 1) * 32, wf[(it + 1) & 1], tid);
    }

    const float* ar = &xf[it & 1][(m0 + lrow) * 32];
    v16bf A = afrag_f32(ar, kbase);
#pragma unroll
    for (int j = 0; j < 4; ++j) {
      v16bf B = bfrag_f32(&wf[it & 1][(j * 16 + lrow) * 32 + kb2]);
      acc[j] = wmma_bf16(A, B, acc[j]);
    }
  }

  // store, pre-activations laid out [t][b][n] for the recurrence
#pragma unroll
  for (int j = 0; j < 4; ++j) {
    int n = j * 16 + lrow;
#pragma unroll
    for (int v = 0; v < 8; ++v) {
      int bt = row0 + m0 + v + mhalf;
      int b  = bt >> 9;          // T = 512
      int t  = bt & 511;
      P[((size_t)t * BATCH + b) * HID + n] = acc[j][v] + bsum[j];
    }
  }
}

// =====================================================================
// Kernel 2: one RNN layer recurrence, fused with next layer's input GEMM.
//   h_t = tanh(P[t] + h_{t-1} @ W_hh^T)
//   Pn[t] = h_t @ W_ihn^T + b_ihn + b_hhn        (if compute_next)
//   out[b] = h_511 . fc_w + fc_b                 (if out != nullptr)
// =====================================================================
__global__ __launch_bounds__(256) void rnn_layer_kernel(
    const float* __restrict__ P,      // [T][B][H] (biases already folded in)
    float* __restrict__ Pn,           // [T][B][H] next-layer pre, or null
    const float* __restrict__ Whh,    // [H][H]   (h_out, h_in)
    const float* __restrict__ Wihn,   // [H][H]   next layer, or null
    const float* __restrict__ bihn, const float* __restrict__ bhhn,
    const float* __restrict__ fcw,  const float* __restrict__ fcb,
    float* __restrict__ out,          // [B] or null
    int compute_next)
{
  __shared__ __align__(16) unsigned char smem[48 * 1024];
  __bf16* hA  = (__bf16*)smem;              // 2 x (128*64) bf16 = 32 KB
  __bf16* whh = (__bf16*)(smem + 32768);    // 64*64 bf16 = 8 KB   [n][k]
  __bf16* wih = (__bf16*)(smem + 40960);    // 64*64 bf16 = 8 KB   [n][k]

  const int tid   = threadIdx.x;
  const int lane  = tid & 31;
  const int wave  = tid >> 5;
  const int lrow  = lane & 15;
  const int kbase = (lane < 16) ? 0 : 16;
  const int kb2   = (lane < 16) ? 0 : 16;
  const int mhalf = (lane < 16) ? 0 : 8;
  const int m0    = wave * 16;

  for (int i = tid; i < HID * HID; i += 256) whh[i] = (__bf16)Whh[i];
  if (compute_next)
    for (int i = tid; i < HID * HID; i += 256) wih[i] = (__bf16)Wihn[i];
  for (int i = tid; i < BATCH * HID; i += 256) hA[i] = (__bf16)0.0f;  // h_{-1}=0

  float bsn[4] = {0.f, 0.f, 0.f, 0.f};
  if (compute_next) {
#pragma unroll
    for (int j = 0; j < 4; ++j) {
      int n = j * 16 + lrow;
      bsn[j] = bihn[n] + bhhn[n];
    }
  }
  __syncthreads();

  int cur = 0;
  // A fragments of h_{t-1} (zeros initially)
  const __bf16* hb = hA + (m0 + lrow) * HID;
  v16bf A0 = cat8(*(const v8bf*)(hb + kbase),      *(const v8bf*)(hb + kbase + 8));
  v16bf A1 = cat8(*(const v8bf*)(hb + 32 + kbase), *(const v8bf*)(hb + 32 + kbase + 8));

  v8f hl[4];  // f32 copy of this wave's h tile (for the FC head)

  for (int t = 0; t < SEQ; ++t) {
    // pull the next step's 32 KB pre-activation slab toward the WGP
    if (t + 1 < SEQ)
      __builtin_prefetch(P + ((size_t)(t + 1) * BATCH) * HID + tid * 32, 0, 1);

    const int nxt = cur ^ 1;
#pragma unroll
    for (int j = 0; j < 4; ++j) {
      const int n = j * 16 + lrow;
      v8f c;
#pragma unroll
      for (int v = 0; v < 8; ++v) {
        int b = m0 + v + mhalf;
        c[v] = P[((size_t)t * BATCH + b) * HID + n];
      }
      const __bf16* wr = &whh[n * HID + kb2];
      v16bf B0 = *(const v16bf*)wr;
      v16bf B1 = *(const v16bf*)(wr + 32);
      c = wmma_bf16(A0, B0, c);
      c = wmma_bf16(A1, B1, c);
#pragma unroll
      for (int v = 0; v < 8; ++v) c[v] = tanhf(c[v]);
      hl[j] = c;
      // publish h_t (bf16) for next step's A operand
#pragma unroll
      for (int v = 0; v < 8; ++v)
        hA[nxt * (BATCH * HID) + (m0 + v + mhalf) * HID + n] = (__bf16)c[v];
    }
    __syncthreads();
    cur = nxt;

    if (compute_next || t < SEQ - 1) {
      const __bf16* hc = hA + cur * (BATCH * HID) + (m0 + lrow) * HID;
      A0 = cat8(*(const v8bf*)(hc + kbase),      *(const v8bf*)(hc + kbase + 8));
      A1 = cat8(*(const v8bf*)(hc + 32 + kbase), *(const v8bf*)(hc + 32 + kbase + 8));
      if (compute_next) {
        // Pn[t] = h_t @ Wihn^T + biases (reuses the A fragments just loaded)
#pragma unroll
        for (int j = 0; j < 4; ++j) {
          const int n = j * 16 + lrow;
          v8f cn = {};
#pragma unroll
          for (int v = 0; v < 8; ++v) cn[v] = bsn[j];
          const __bf16* wr = &wih[n * HID + kb2];
          cn = wmma_bf16(A0, *(const v16bf*)wr, cn);
          cn = wmma_bf16(A1, *(const v16bf*)(wr + 32), cn);
#pragma unroll
          for (int v = 0; v < 8; ++v) {
            int b = m0 + v + mhalf;
            Pn[((size_t)t * BATCH + b) * HID + n] = cn[v];
          }
        }
      }
    }
  }

  // FC head: out[b] = h_511 . fc_w + fc_b (h kept in f32 registers)
  if (out) {
    float* hF = (float*)smem;  // reuse LDS (all hA traffic done)
#pragma unroll
    for (int j = 0; j < 4; ++j)
#pragma unroll
      for (int v = 0; v < 8; ++v)
        hF[(m0 + v + mhalf) * HID + (j * 16 + lrow)] = hl[j][v];
    __syncthreads();
    if (tid < BATCH) {
      float s = fcb[0];
      for (int n = 0; n < HID; ++n) s += hF[tid * HID + n] * fcw[n];
      out[tid] = s;
    }
  }
}

// =====================================================================
extern "C" void kernel_launch(void* const* d_in, const int* in_sizes, int n_in,
                              void* d_out, int out_size, void* d_ws, size_t ws_size,
                              hipStream_t stream) {
  (void)in_sizes; (void)n_in; (void)out_size; (void)ws_size;
  const float* x     = (const float*)d_in[0];
  // d_in[1] = fixation_num (unused)
  const float* W_ih0 = (const float*)d_in[2];
  const float* W_hh0 = (const float*)d_in[3];
  const float* b_ih0 = (const float*)d_in[4];
  const float* b_hh0 = (const float*)d_in[5];
  const float* W_ih1 = (const float*)d_in[6];
  const float* W_hh1 = (const float*)d_in[7];
  const float* b_ih1 = (const float*)d_in[8];
  const float* b_hh1 = (const float*)d_in[9];
  const float* W_ih2 = (const float*)d_in[10];
  const float* W_hh2 = (const float*)d_in[11];
  const float* b_ih2 = (const float*)d_in[12];
  const float* b_hh2 = (const float*)d_in[13];
  const float* fc_w  = (const float*)d_in[14];
  const float* fc_b  = (const float*)d_in[15];

  float* P0 = (float*)d_ws;                       // 65536*64 f32 = 16 MB
  float* P1 = P0 + (size_t)BATCH * SEQ * HID;     // second 16 MB

  // Layer 0 input GEMM (memory-bound: 192 MB of x @ 23.3 TB/s)
  pre0_gemm_kernel<<<512, 256, 0, stream>>>(x, W_ih0, b_ih0, b_hh0, P0);
  // Recurrence layer 0, fused with layer-1 input GEMM
  rnn_layer_kernel<<<1, 256, 0, stream>>>(P0, P1, W_hh0, W_ih1, b_ih1, b_hh1,
                                          nullptr, nullptr, nullptr, 1);
  // Recurrence layer 1, fused with layer-2 input GEMM
  rnn_layer_kernel<<<1, 256, 0, stream>>>(P1, P0, W_hh1, W_ih2, b_ih2, b_hh2,
                                          nullptr, nullptr, nullptr, 1);
  // Recurrence layer 2 + FC head
  rnn_layer_kernel<<<1, 256, 0, stream>>>(P0, nullptr, W_hh2, nullptr, nullptr,
                                          nullptr, fc_w, fc_b, (float*)d_out, 0);
}
